// NMTEncoder_30958124269570
// MI455X (gfx1250) — matrix operand order repaired and checked by reading, not compiled, not run on hardware
//
#include <hip/hip_runtime.h>
#include <cstdint>
#include <cstddef>

#define VOCAB 32000
#define EMB   512
#define HSZ   1024
#define BB    128
#define SS    256
#define K3H   3072           // 3*HSZ
#define KTOT  1536           // EMB + HSZ
#define NKC   48             // KTOT / 32 (bf16 WMMA K per instruction)
#define NCT   192            // K3H / 16 (column tiles of packed W)

typedef __attribute__((ext_vector_type(16))) __bf16 v16bf;
typedef __attribute__((ext_vector_type(8)))  float  v8f;
typedef __attribute__((ext_vector_type(4)))  int    int4v;

union Frag { unsigned u[8]; int4v i4[2]; v16bf v; };

__device__ __forceinline__ unsigned short f2bf(float f) {
    unsigned u = __builtin_bit_cast(unsigned, f);
    unsigned r = u + 0x7FFFu + ((u >> 16) & 1u);   // round-to-nearest-even
    return (unsigned short)(r >> 16);
}

__device__ __forceinline__ float fast_sigmoid(float x) {
    return 1.0f / (1.0f + __expf(-x));
}
__device__ __forceinline__ float fast_tanh(float x) {
    float e = __expf(-2.0f * x);                   // v_exp_f32 path
    return (1.0f - e) / (1.0f + e);
}

// ---------------------------------------------------------------------------
// Pack [W_ih ; W_hh] (fp32, 1536 x 3072) into bf16 WMMA B-fragment order:
// element (kc, ct, lane, j): k = kc*32 + (lane/16)*16 + j, n = ct*16 + lane%16
// -> each lane's fragment is 16 contiguous bf16 (two b128 loads).
// ---------------------------------------------------------------------------
__global__ void pack_w_kernel(const float* __restrict__ Wih,
                              const float* __restrict__ Whh,
                              unsigned short* __restrict__ Wpk) {
    int p = blockIdx.x * blockDim.x + threadIdx.x;
    if (p >= KTOT * K3H) return;
    int j    = p & 15;
    int lane = (p >> 4) & 31;
    int ct   = (p >> 9) % NCT;
    int kc   = p / (512 * NCT);
    int k    = kc * 32 + ((lane >> 4) << 4) + j;
    int n    = ct * 16 + (lane & 15);
    float v  = (k < EMB) ? Wih[(size_t)k * K3H + n]
                         : Whh[(size_t)(k - EMB) * K3H + n];
    Wpk[p] = f2bf(v);
}

// ---------------------------------------------------------------------------
// Gather + convert embeddings once: Ebf[s][b][k] (bf16), padding id 0 -> 0.
// ---------------------------------------------------------------------------
__global__ void pack_emb_kernel(const int* __restrict__ xsrc,
                                const float* __restrict__ embt,
                                unsigned short* __restrict__ Ebf) {
    size_t t = (size_t)blockIdx.x * blockDim.x + threadIdx.x;
    if (t >= (size_t)SS * BB * EMB) return;
    int k = (int)(t % EMB);
    int b = (int)((t / EMB) % BB);
    int s = (int)(t / ((size_t)EMB * BB));
    int tok = xsrc[b * SS + s];
    float v = (tok != 0) ? embt[(size_t)tok * EMB + k] : 0.0f;
    Ebf[t] = f2bf(v);
}

__global__ void zero_h_kernel(float* __restrict__ hf,
                              unsigned short* __restrict__ hbf) {
    int i = blockIdx.x * blockDim.x + threadIdx.x;
    if (i < BB * HSZ) { hf[i] = 0.0f; hbf[i] = 0; }
}

// ---------------------------------------------------------------------------
// One GRU timestep: gates = [emb_s , h_prev] @ [W_ih ; W_hh]  (+ biases).
// grid = (32, 2), block = 128 (4 waves). Wave w: rows [mblk*64+16w, +16);
// block covers TWO 16-col tiles, so each A fragment feeds 6 WMMAs.
// K loop is split emb/h so each phase is straight-line loads + WMMA.
// ---------------------------------------------------------------------------
__global__ void __launch_bounds__(128) gru_step_kernel(
    const unsigned short* __restrict__ Ebf,      // [S][B][EMB] bf16
    const unsigned short* __restrict__ Wpk,      // packed weights
    const float* __restrict__ bih,
    const float* __restrict__ bhh,
    const float* __restrict__ hprev_f,           // [B][H] fp32 (exact carry)
    const unsigned short* __restrict__ hprev_bf, // [B][H] bf16 (GEMM operand)
    float*       __restrict__ hnext_f,
    unsigned short* __restrict__ hnext_bf,
    float*       __restrict__ out,
    int s)
{
    const int tid  = threadIdx.x;
    const int lane = tid & 31;
    const int w    = tid >> 5;                 // wave 0..3
    const int cth0 = blockIdx.x * 2;           // first of two column tiles
    const int mblk = blockIdx.y;               // 0..1

    // A-fragment addressing (contiguous K order, matches packed B): m = lane%16,
    // lanes>=16 take the +16 K half; fragment = [k0..k0+15].
    const int mrow = mblk * 64 + w * 16 + (lane & 15);
    const int klo  = (lane >> 4) << 4;         // 0 or 16

    const unsigned short* Arow_e = Ebf + ((size_t)s * BB + mrow) * EMB + klo;
    const unsigned short* Arow_h = hprev_bf + (size_t)mrow * HSZ + klo;

    // B fragment bases for the 6 (gate, tile) slabs; advance by one kc-slab.
    const size_t lane16 = (size_t)lane * 16;
    const unsigned short* Br0 = Wpk + ((size_t)(        cth0    ) * 32) * 16 + lane16;
    const size_t kcstride = (size_t)NCT * 32 * 16;     // elements per kc slab

    v8f cr0 = {}, cz0 = {}, cxn0 = {}, chn0 = {};
    v8f cr1 = {}, cz1 = {}, cxn1 = {}, chn1 = {};

    // Byte-element offsets (in ushort units) of the 6 fragments from Br0.
    const size_t o_r1 = (size_t)32 * 16;                       // cth0+1
    const size_t o_z0 = (size_t)64  * 32 * 16;                 // z slab
    const size_t o_z1 = o_z0 + o_r1;
    const size_t o_n0 = (size_t)128 * 32 * 16;                 // n slab
    const size_t o_n1 = o_n0 + o_r1;

    // ---------------- phase 1: embedding slab (kc = 0..15) ----------------
    {
        const unsigned short* Bp = Br0;
        #pragma unroll 2
        for (int kc = 0; kc < 16; ++kc) {
            Frag a;
            {
                const int4v* pa = (const int4v*)(Arow_e + kc * 32);
                a.i4[0] = pa[0];
                a.i4[1] = pa[1];
            }
            Frag br0, bz0, bn0, br1, bz1, bn1;
            br0.i4[0] = ((const int4v*)(Bp        ))[0]; br0.i4[1] = ((const int4v*)(Bp        ))[1];
            br1.i4[0] = ((const int4v*)(Bp + o_r1))[0]; br1.i4[1] = ((const int4v*)(Bp + o_r1))[1];
            bz0.i4[0] = ((const int4v*)(Bp + o_z0))[0]; bz0.i4[1] = ((const int4v*)(Bp + o_z0))[1];
            bz1.i4[0] = ((const int4v*)(Bp + o_z1))[0]; bz1.i4[1] = ((const int4v*)(Bp + o_z1))[1];
            bn0.i4[0] = ((const int4v*)(Bp + o_n0))[0]; bn0.i4[1] = ((const int4v*)(Bp + o_n0))[1];
            bn1.i4[0] = ((const int4v*)(Bp + o_n1))[0]; bn1.i4[1] = ((const int4v*)(Bp + o_n1))[1];

            cr0  = __builtin_amdgcn_wmma_f32_16x16x32_bf16(false, a.v, false, br0.v, (short)0, cr0,  false, false);
            cr1  = __builtin_amdgcn_wmma_f32_16x16x32_bf16(false, a.v, false, br1.v, (short)0, cr1,  false, false);
            cz0  = __builtin_amdgcn_wmma_f32_16x16x32_bf16(false, a.v, false, bz0.v, (short)0, cz0,  false, false);
            cz1  = __builtin_amdgcn_wmma_f32_16x16x32_bf16(false, a.v, false, bz1.v, (short)0, cz1,  false, false);
            cxn0 = __builtin_amdgcn_wmma_f32_16x16x32_bf16(false, a.v, false, bn0.v, (short)0, cxn0, false, false);
            cxn1 = __builtin_amdgcn_wmma_f32_16x16x32_bf16(false, a.v, false, bn1.v, (short)0, cxn1, false, false);

            Bp += kcstride;
        }
    }

    // ---------------- phase 2: hidden-state slab (kc = 16..47) -------------
    {
        const unsigned short* Bp = Br0 + (size_t)16 * kcstride;
        #pragma unroll 2
        for (int kc = 0; kc < 32; ++kc) {
            Frag a;
            {
                const int4v* pa = (const int4v*)(Arow_h + kc * 32);
                a.i4[0] = pa[0];
                a.i4[1] = pa[1];
            }
            Frag br0, bz0, bn0, br1, bz1, bn1;
            br0.i4[0] = ((const int4v*)(Bp        ))[0]; br0.i4[1] = ((const int4v*)(Bp        ))[1];
            br1.i4[0] = ((const int4v*)(Bp + o_r1))[0]; br1.i4[1] = ((const int4v*)(Bp + o_r1))[1];
            bz0.i4[0] = ((const int4v*)(Bp + o_z0))[0]; bz0.i4[1] = ((const int4v*)(Bp + o_z0))[1];
            bz1.i4[0] = ((const int4v*)(Bp + o_z1))[0]; bz1.i4[1] = ((const int4v*)(Bp + o_z1))[1];
            bn0.i4[0] = ((const int4v*)(Bp + o_n0))[0]; bn0.i4[1] = ((const int4v*)(Bp + o_n0))[1];
            bn1.i4[0] = ((const int4v*)(Bp + o_n1))[0]; bn1.i4[1] = ((const int4v*)(Bp + o_n1))[1];

            cr0  = __builtin_amdgcn_wmma_f32_16x16x32_bf16(false, a.v, false, br0.v, (short)0, cr0,  false, false);
            cr1  = __builtin_amdgcn_wmma_f32_16x16x32_bf16(false, a.v, false, br1.v, (short)0, cr1,  false, false);
            cz0  = __builtin_amdgcn_wmma_f32_16x16x32_bf16(false, a.v, false, bz0.v, (short)0, cz0,  false, false);
            cz1  = __builtin_amdgcn_wmma_f32_16x16x32_bf16(false, a.v, false, bz1.v, (short)0, cz1,  false, false);
            chn0 = __builtin_amdgcn_wmma_f32_16x16x32_bf16(false, a.v, false, bn0.v, (short)0, chn0, false, false);
            chn1 = __builtin_amdgcn_wmma_f32_16x16x32_bf16(false, a.v, false, bn1.v, (short)0, chn1, false, false);

            Bp += kcstride;
        }
    }

    // ---- epilogue: biases, gates, carry update (fp32), bf16 shadow ----
    const int mb = mblk * 64 + w * 16 + ((lane >> 4) << 3);  // C/D: M = v (+8 hi)

    #pragma unroll
    for (int t = 0; t < 2; ++t) {
        const v8f& cr  = t ? cr1  : cr0;
        const v8f& cz  = t ? cz1  : cz0;
        const v8f& cxn = t ? cxn1 : cxn0;
        const v8f& chn = t ? chn1 : chn0;
        const int  j   = (cth0 + t) * 16 + (lane & 15);
        const float br_ = bih[j]           + bhh[j];
        const float bz_ = bih[HSZ + j]     + bhh[HSZ + j];
        const float bxn = bih[2 * HSZ + j];
        const float bhn = bhh[2 * HSZ + j];
        #pragma unroll
        for (int v = 0; v < 8; ++v) {
            int   m    = mb + v;
            float r    = fast_sigmoid(cr[v] + br_);
            float z    = fast_sigmoid(cz[v] + bz_);
            float n    = fast_tanh(cxn[v] + bxn + r * (chn[v] + bhn));
            float hold = hprev_f[(size_t)m * HSZ + j];
            float hnew = (1.0f - z) * n + z * hold;
            hnext_f[(size_t)m * HSZ + j]        = hnew;
            hnext_bf[(size_t)m * HSZ + j]       = f2bf(hnew);
            out[((size_t)m * SS + s) * HSZ + j] = hnew;
        }
    }
}

extern "C" void kernel_launch(void* const* d_in, const int* in_sizes, int n_in,
                              void* d_out, int out_size, void* d_ws, size_t ws_size,
                              hipStream_t stream) {
    const int*   xsrc = (const int*)d_in[0];
    // d_in[1] = x_lengths: unused (max length == S statically in the reference)
    const float* embt = (const float*)d_in[2];
    const float* Wih  = (const float*)d_in[3];
    const float* Whh  = (const float*)d_in[4];
    const float* bih  = (const float*)d_in[5];
    const float* bhh  = (const float*)d_in[6];
    float*       out  = (float*)d_out;

    // ---- workspace layout ----
    char* ws = (char*)d_ws;
    unsigned short* Wpk = (unsigned short*)ws;                 //  9.4 MB
    ws += (size_t)KTOT * K3H * sizeof(unsigned short);
    unsigned short* Ebf = (unsigned short*)ws;                 // 33.6 MB
    ws += (size_t)SS * BB * EMB * sizeof(unsigned short);
    float* hf0 = (float*)ws;  ws += (size_t)BB * HSZ * sizeof(float);
    float* hf1 = (float*)ws;  ws += (size_t)BB * HSZ * sizeof(float);
    unsigned short* hb0 = (unsigned short*)ws; ws += (size_t)BB * HSZ * sizeof(unsigned short);
    unsigned short* hb1 = (unsigned short*)ws;

    {
        int total = KTOT * K3H;
        pack_w_kernel<<<(total + 255) / 256, 256, 0, stream>>>(Wih, Whh, Wpk);
    }
    {
        size_t total = (size_t)SS * BB * EMB;
        pack_emb_kernel<<<(unsigned)((total + 255) / 256), 256, 0, stream>>>(xsrc, embt, Ebf);
    }
    zero_h_kernel<<<(BB * HSZ + 255) / 256, 256, 0, stream>>>(hf0, hb0);

    float*          hf[2] = { hf0, hf1 };
    unsigned short* hb[2] = { hb0, hb1 };
    for (int s = 0; s < SS; ++s) {
        gru_step_kernel<<<dim3(32, 2), 128, 0, stream>>>(
            Ebf, Wpk, bih, bhh,
            hf[s & 1], hb[s & 1],
            hf[(s + 1) & 1], hb[(s + 1) & 1],
            out, s);
    }

    // h_last lives in hf[S & 1] (== hf0 for even S); append after [B,S,H]
    hipMemcpyAsync(out + (size_t)BB * SS * HSZ, hf[SS & 1],
                   (size_t)BB * HSZ * sizeof(float),
                   hipMemcpyDeviceToDevice, stream);
}
